// MaskedAttention_82729660056348
// MI455X (gfx1250) — matrix operand order repaired and checked by reading, not compiled, not run on hardware
//
#include <hip/hip_runtime.h>
#include <hip/hip_bf16.h>

typedef __attribute__((ext_vector_type(16))) _Float16 v16h;
typedef __attribute__((ext_vector_type(8)))  float    v8f;
typedef __attribute__((ext_vector_type(4)))  unsigned v4u;
typedef __attribute__((ext_vector_type(8)))  int      v8i;
typedef __attribute__((ext_vector_type(4)))  int      v4i;

#define Bsz 16
#define Tsz 2048
#define Csz 1024
#define Dsz 64

#if defined(__has_builtin)
#  if __has_builtin(__builtin_amdgcn_tensor_load_to_lds) && __has_builtin(__builtin_amdgcn_s_wait_tensorcnt)
#    define USE_TDM 1
#  endif
#endif
#ifndef USE_TDM
#  define USE_TDM 0
#endif

__device__ inline v8f wmma_f16(v16h a, v16h b, v8f c) {
    return __builtin_amdgcn_wmma_f32_16x16x32_f16(
        false, a, false, b, (short)0, c, false, false);
}

// A-matrix 16x32 f16: lane r=lane&15 holds row M=r.
// lanes 0-15: K=0..15; lanes 16-31: K=16..31. Two 16B loads.
__device__ inline v16h load_a16(const _Float16* base, int ldr, int lane) {
    const int r     = lane & 15;
    const int khalf = (lane >> 4) * 16;
    v16h a;
    *(uint4*)&a                 = *(const uint4*)(base + (size_t)r * ldr + khalf);
    *(uint4*)(((char*)&a) + 16) = *(const uint4*)(base + (size_t)r * ldr + 8 + khalf);
    return a;
}

// B-matrix 32x16 f16: n=lane&15, lanes 0-15 hold K=0..15, lanes 16-31 K=16..31.
// Source stored K-major per n-row: B(k,n) = base[n*ldr + k] -> per-lane contiguous,
// so the whole operand is two 16B loads per lane.
__device__ inline v16h load_b_kmajor(const _Float16* base, int ldr, int lane) {
    const int n    = lane & 15;
    const int kblk = (lane >> 4) * 16;
    v16h vb;
    *(uint4*)&vb                 = *(const uint4*)(base + (size_t)n * ldr + kblk);
    *(uint4*)(((char*)&vb) + 16) = *(const uint4*)(base + (size_t)n * ldr + kblk + 8);
    return vb;
}

// low 32 bits of a flat pointer to LDS == LDS byte offset (ISA 10.2 aperture rule)
__device__ inline unsigned lds_off(const void* p) {
    return (unsigned)(unsigned long long)p;
}

// ---------------- Kernel 0: weight transpose-convert  W[1024][64] f32 -> Wt[64][1024] f16
__global__ __launch_bounds__(256)
void wcvt_kernel(const float* __restrict__ Wq, const float* __restrict__ Wk,
                 const float* __restrict__ Wv, _Float16* __restrict__ Wt)
{
    const int mat = blockIdx.x >> 6;   // 0..2
    const int d   = blockIdx.x & 63;
    const float* W = (mat == 0) ? Wq : (mat == 1) ? Wk : Wv;
    _Float16* dst  = Wt + ((size_t)mat * Dsz + d) * Csz;
    const int c = threadIdx.x * 4;
#pragma unroll
    for (int i = 0; i < 4; ++i)
        dst[c + i] = (_Float16)W[(size_t)(c + i) * Dsz + d];
}

// ---------------- Kernel 1: QKV projection ----------------
// grid = B*T/16 blocks, 128 threads (4 waves); wave w owns d-slice [16w,16w+16).
__global__ __launch_bounds__(128)
void qkv_kernel(const float* __restrict__ x,
                const _Float16* __restrict__ Wt,  // [3][64][1024] f16 (q,k,v)
                _Float16* __restrict__ Qh,        // [B][T][64], pre-scaled 1/sqrt(C)
                _Float16* __restrict__ Kh,        // [B][T][64]
                _Float16* __restrict__ Vt)        // [B][64][T] (transposed)
{
    __shared__ __align__(16) _Float16 sx[16 * 32];

    const int tid  = threadIdx.x;
    const int lane = tid & 31;
    const int wave = tid >> 5;
    const int b    = blockIdx.x >> 7;
    const int t0   = (blockIdx.x & 127) << 4;

    const float* xrow = x + ((size_t)b * Tsz + t0) * Csz;
    const _Float16* wq = Wt + (size_t)(0 * Dsz + wave * 16) * Csz;
    const _Float16* wk = Wt + (size_t)(1 * Dsz + wave * 16) * Csz;
    const _Float16* wv = Wt + (size_t)(2 * Dsz + wave * 16) * Csz;

    v8f cq = {}; v8f ck = {}; v8f cv = {};

    const int xr = tid >> 3;
    const int xc = (tid & 7) * 4;

    for (int kk = 0; kk < Csz; kk += 32) {
        const float4 xv = *(const float4*)(xrow + (size_t)xr * Csz + kk + xc);
        __syncthreads();
        sx[xr * 32 + xc + 0] = (_Float16)xv.x;
        sx[xr * 32 + xc + 1] = (_Float16)xv.y;
        sx[xr * 32 + xc + 2] = (_Float16)xv.z;
        sx[xr * 32 + xc + 3] = (_Float16)xv.w;
        __syncthreads();

        const v16h a = load_a16(sx, 32, lane);
        cq = wmma_f16(a, load_b_kmajor(wq + kk, Csz, lane), cq);
        ck = wmma_f16(a, load_b_kmajor(wk + kk, Csz, lane), ck);
        cv = wmma_f16(a, load_b_kmajor(wv + kk, Csz, lane), cv);
    }

    const int h = lane >> 4, n = lane & 15;
#pragma unroll
    for (int v = 0; v < 8; ++v) {
        const int t = t0 + v + 8 * h;
        const int d = wave * 16 + n;
        Qh[((size_t)b * Tsz + t) * Dsz + d] = (_Float16)(cq[v] * 0.03125f);
        Kh[((size_t)b * Tsz + t) * Dsz + d] = (_Float16)ck[v];
        Vt[((size_t)b * Dsz + d) * Tsz + t] = (_Float16)cv[v];
    }
}

// ---------------- Kernel 2: causal flash attention ----------------
// grid = B*T/64 blocks, 128 threads (4 waves); wave w owns queries [qbase+16w, +16).
__global__ __launch_bounds__(128)
void attn_kernel(const _Float16* __restrict__ Qh,
                 const _Float16* __restrict__ Kh,
                 const _Float16* __restrict__ Vt,
                 float* __restrict__ out)
{
    __shared__ __align__(16) _Float16 sK[32 * 64];   // [key][d]  (contiguous copy of 32 K rows)
    __shared__ __align__(16) _Float16 sVt[64 * 32];  // [d][key]
    __shared__ __align__(16) _Float16 sP[4][16 * 32];

    const int tid   = threadIdx.x;
    const int lane  = tid & 31;
    const int wave  = tid >> 5;
    const int b     = blockIdx.x >> 5;
    const int qbase = (blockIdx.x & 31) << 6;
    const int qrow0 = qbase + wave * 16;

    const _Float16* qptr = Qh + ((size_t)b * Tsz + qrow0) * Dsz;
    const v16h aq0 = load_a16(qptr,      Dsz, lane);
    const v16h aq1 = load_a16(qptr + 32, Dsz, lane);

    v8f o0 = {}, o1 = {}, o2 = {}, o3 = {};
    float m[8], l[8];
#pragma unroll
    for (int v = 0; v < 8; ++v) { m[v] = -__builtin_inff(); l[v] = 0.f; }

    const int h = lane >> 4, n = lane & 15;
    const int nkb = (qbase + 64) >> 5;

    for (int kb32 = 0; kb32 < nkb; ++kb32) {
        const int kb = kb32 * 32;
        __syncthreads();                       // everyone done with sK/sVt

#if USE_TDM
        if (tid < 32) {                        // wave 0 drives the Tensor Data Mover
            // K tile: 32 rows x 64 d = one contiguous 4KB run (1-D tile)
            const unsigned long long ka =
                (unsigned long long)(Kh + ((size_t)b * Tsz + kb) * Dsz);
            const v4u g0k = { 1u, lds_off(sK), (unsigned)ka,
                              (unsigned)((ka >> 32) & 0x01ffffffu) | 0x80000000u };
            const v8i g1k = { 0x00010000,            // data_size=2B, mask=0
                              (int)(2048u << 16),    // tensor_dim0 = 2048 elems
                              (int)(1u << 16),       // tensor_dim1 = 1
                              (int)(2048u << 16),    // tile_dim0   = 2048
                              1,                     // tile_dim1   = 1
                              2048, 0, 0 };          // dim0 stride
            // V tile: 64 rows (d) x 32 keys, row stride T (2-D tile)
            const unsigned long long va =
                (unsigned long long)(Vt + (size_t)b * Dsz * Tsz + kb);
            const v4u g0v = { 1u, lds_off(sVt), (unsigned)va,
                              (unsigned)((va >> 32) & 0x01ffffffu) | 0x80000000u };
            const v8i g1v = { 0x00010000,
                              (int)(32u << 16),      // tensor_dim0 = 32
                              (int)(64u << 16),      // tensor_dim1 = 64
                              (int)(32u << 16),      // tile_dim0   = 32
                              64,                    // tile_dim1   = 64
                              (int)Tsz, 0, 0 };      // dim0 stride = T
            const v4i z4 = { 0, 0, 0, 0 };
            const v8i z8 = { 0, 0, 0, 0, 0, 0, 0, 0 };
            __builtin_amdgcn_tensor_load_to_lds(g0k, g1k, z4, z4, z8, 0);
            __builtin_amdgcn_tensor_load_to_lds(g0v, g1v, z4, z4, z8, 0);
            __builtin_amdgcn_s_wait_tensorcnt(0);
        }
#else
#pragma unroll
        for (int e = tid; e < 256; e += 128)   // K: contiguous 4KB
            ((uint4*)sK)[e] = ((const uint4*)(Kh + ((size_t)b * Tsz + kb) * Dsz))[e];
#pragma unroll
        for (int e = tid; e < 256; e += 128) { // V: 64 rows of 64B
            const int d = e >> 2, off = (e & 3) * 8;
            ((uint4*)sVt)[e] = *(const uint4*)(Vt + ((size_t)b * Dsz + d) * Tsz + kb + off);
        }
#endif
        __syncthreads();

        if (kb <= qrow0 + 15) {                // wave-uniform: EXEC all-ones for WMMA
            // S = Q*K^T: B(k=d', n=key) = sK[key][d] -> K-major per n, 2x16B loads
            v8f s0 = {}, s1 = {};
            s0 = wmma_f16(aq0, load_b_kmajor(sK,                 Dsz, lane), s0);
            s0 = wmma_f16(aq1, load_b_kmajor(sK + 32,            Dsz, lane), s0);
            s1 = wmma_f16(aq0, load_b_kmajor(sK + 16 * Dsz,      Dsz, lane), s1);
            s1 = wmma_f16(aq1, load_b_kmajor(sK + 16 * Dsz + 32, Dsz, lane), s1);

            _Float16* sPw = &sP[wave][0];
#pragma unroll
            for (int v = 0; v < 8; ++v) {
                const int q  = qrow0 + v + 8 * h;
                const int k0 = kb + n;
                float sv0 = (k0      <= q) ? s0[v] : -__builtin_inff();
                float sv1 = (k0 + 16 <= q) ? s1[v] : -__builtin_inff();
                float r = fmaxf(sv0, sv1);
                r = fmaxf(r, __shfl_xor(r, 1, 32));
                r = fmaxf(r, __shfl_xor(r, 2, 32));
                r = fmaxf(r, __shfl_xor(r, 4, 32));
                r = fmaxf(r, __shfl_xor(r, 8, 32));
                const float mn   = fmaxf(m[v], r);
                const bool  minf = (mn == -__builtin_inff());
                const float alpha = minf ? 1.f : __expf(m[v] - mn);
                const float p0 = minf ? 0.f : __expf(sv0 - mn);
                const float p1 = minf ? 0.f : __expf(sv1 - mn);
                float ps = p0 + p1;
                ps += __shfl_xor(ps, 1, 32);
                ps += __shfl_xor(ps, 2, 32);
                ps += __shfl_xor(ps, 4, 32);
                ps += __shfl_xor(ps, 8, 32);
                l[v] = l[v] * alpha + ps;
                m[v] = mn;
                o0[v] *= alpha; o1[v] *= alpha; o2[v] *= alpha; o3[v] *= alpha;
                sPw[(v + 8 * h) * 32 + n]      = (_Float16)p0;
                sPw[(v + 8 * h) * 32 + 16 + n] = (_Float16)p1;
            }
            const v16h ap = load_a16(sPw, 32, lane);  // same-wave LDS RAW: in-order
            // O += P*V: B(k=key, n=d) = sVt[d][key] -> K-major per n
            o0 = wmma_f16(ap, load_b_kmajor(sVt,           32, lane), o0);
            o1 = wmma_f16(ap, load_b_kmajor(sVt + 16 * 32, 32, lane), o1);
            o2 = wmma_f16(ap, load_b_kmajor(sVt + 32 * 32, 32, lane), o2);
            o3 = wmma_f16(ap, load_b_kmajor(sVt + 48 * 32, 32, lane), o3);
        }
    }

#pragma unroll
    for (int v = 0; v < 8; ++v) {
        const float inv = 1.0f / l[v];
        const int q = qrow0 + v + 8 * h;
        float* op = out + ((size_t)b * Tsz + q) * Dsz + n;
        op[0]  = o0[v] * inv;
        op[16] = o1[v] * inv;
        op[32] = o2[v] * inv;
        op[48] = o3[v] * inv;
    }
}

extern "C" void kernel_launch(void* const* d_in, const int* in_sizes, int n_in,
                              void* d_out, int out_size, void* d_ws, size_t ws_size,
                              hipStream_t stream) {
    const float* x  = (const float*)d_in[0];
    const float* Wq = (const float*)d_in[1];
    const float* Wk = (const float*)d_in[2];
    const float* Wv = (const float*)d_in[3];
    float* out = (float*)d_out;

    const size_t qkv_elems = (size_t)Bsz * Tsz * Dsz;          // 2M halves each
    const size_t wt_elems  = (size_t)3 * Dsz * Csz;            // 192K halves
    if (ws_size < (3 * qkv_elems + wt_elems) * sizeof(_Float16)) return;
    _Float16* Qh  = (_Float16*)d_ws;
    _Float16* Kh  = Qh + qkv_elems;
    _Float16* Vt  = Kh + qkv_elems;
    _Float16* Wth = Vt + qkv_elems;

    wcvt_kernel<<<dim3(3 * Dsz), dim3(256), 0, stream>>>(Wq, Wk, Wv, Wth);
    qkv_kernel<<<dim3(Bsz * (Tsz / 16)), dim3(128), 0, stream>>>(x, Wth, Qh, Kh, Vt);
    attn_kernel<<<dim3(Bsz * (Tsz / 64)), dim3(128), 0, stream>>>(Qh, Kh, Vt, out);
}
